// CrossViewSwapAttention_9423158247629
// MI455X (gfx1250) — matrix operand order repaired and checked by reading, not compiled, not run on hardware
//
#include <hip/hip_runtime.h>
#include <hip/hip_bf16.h>
#include <math.h>

typedef _Float16 h16;
typedef __attribute__((ext_vector_type(16))) _Float16 v16h;
typedef __attribute__((ext_vector_type(8)))  _Float16 v8h;
typedef __attribute__((ext_vector_type(8)))  float    v8f;

#define NCAM   6
#define CDIM   128
#define FHW    48
#define HWF    2304      /* 48*48 */
#define HB     96
#define HWB    9216      /* 96*96 */
#define NQTOK  55296     /* 36 windows * 6 cams * 256 */
#define NKTOK  13824     /* 36 windows * 6 cams * 64  (== 6*2304 spatial) */
#define HEADS  4
#define DH     32

// ---------------------------------------------------------------------------
// WMMA fragment helpers (layouts per CDNA5 ISA 7.12.2, 16-bit, K=32).
// Both A and B fragments are two contiguous 16-byte chunks per lane, so we
// load them with 128-bit vector loads.
// ---------------------------------------------------------------------------
__device__ __forceinline__ int laneid() { return threadIdx.x & 31; }

__device__ __forceinline__ v16h combine16(v8h c0, v8h c1) {
  v16h f;
#pragma unroll
  for (int i = 0; i < 8; ++i) { f[i] = c0[i]; f[i + 8] = c1[i]; }
  return f;
}

// A-matrix 16x32: lane 0-15 row=lane holds K {hi..hi+7, 16+hi..23+hi}, hi=0;
// lanes 16-31 hi=8. Two contiguous 8-half chunks at +hi and +hi+16.
__device__ __forceinline__ v16h load_a16(const h16* p, int row0, int k0, int ld) {
  int l = laneid();
  int hi = (l & 16) ? 8 : 0;
  const h16* base = p + (size_t)(row0 + (l & 15)) * ld + k0 + hi;
  return combine16(*(const v8h*)base, *(const v8h*)(base + 16));
}

// B-matrix 32x16 where B[k][n] = p[(n0+n)*ld + k0 + k] (p row-major [N x K]):
// lane = column n; lanes 0-15 K={0..7,16..23}, lanes 16-31 K={8..15,24..31}.
__device__ __forceinline__ v16h load_b16(const h16* p, int n0, int k0, int ld) {
  int l = laneid();
  int lo = (l & 16) ? 8 : 0;
  const h16* base = p + (size_t)(n0 + (l & 15)) * ld + k0 + lo;
  return combine16(*(const v8h*)base, *(const v8h*)(base + 16));
}

__device__ __forceinline__ v8f wmma_f16(v16h a, v16h b, v8f c) {
  return __builtin_amdgcn_wmma_f32_16x16x32_f16(false, a, false, b, (short)0, c,
                                                false, false);
}

// ---------------------------------------------------------------------------
// Generic GEMM:  C[M x N] = A_f16[M x K] @ W_f16[N x K]^T (+ bias)
// One wave -> 32x64 tile (2 A-frags x 4 B-frags, 8 WMMAs / K-step),
// 8 waves/block. M % 256 == 0, N % 64 == 0, K % 32 == 0.
// MODE: 0 = f32 out, 1 = f16 out, 2 = f16 out with exact GELU,
//       3 = f16 out TRANSPOSED (out[col * MT + row], leading dim MT)
// ---------------------------------------------------------------------------
template <int MODE>
__device__ __forceinline__ void store_tile(void* outp, const v8f& c, int mb,
                                           int col, int N, int MT, float bv) {
#pragma unroll
  for (int v = 0; v < 8; ++v) {
    float val = c[v] + bv;
    if (MODE == 0) {
      ((float*)outp)[(size_t)(mb + v) * N + col] = val;
    } else if (MODE == 1) {
      ((h16*)outp)[(size_t)(mb + v) * N + col] = (h16)val;
    } else if (MODE == 2) {
      float g = 0.5f * val * (1.0f + erff(val * 0.70710678118654752f));
      ((h16*)outp)[(size_t)(mb + v) * N + col] = (h16)g;
    } else {
      ((h16*)outp)[(size_t)col * MT + (mb + v)] = (h16)val;
    }
  }
}

template <int MODE>
__global__ __launch_bounds__(256) void gemm_k(const h16* __restrict__ A,
                                              const h16* __restrict__ W,
                                              const float* __restrict__ bias,
                                              void* __restrict__ outp,
                                              int N, int K, int MT) {
  int wave = threadIdx.x >> 5;
  int row0 = (blockIdx.x * 8 + wave) * 32;
  int n0 = blockIdx.y * 64;
  v8f zero = {};
  v8f acc[2][4];
#pragma unroll
  for (int r = 0; r < 2; ++r)
#pragma unroll
    for (int t = 0; t < 4; ++t) acc[r][t] = zero;

  for (int k0 = 0; k0 < K; k0 += 32) {
    if (k0 + 32 < K) {  // wave-uniform: prefetch next K-step (global_prefetch_b8)
      __builtin_prefetch(A + (size_t)(row0 + (laneid() & 15)) * K + k0 + 32, 0, 3);
      __builtin_prefetch(W + (size_t)(n0 + (laneid() & 15)) * K + k0 + 32, 0, 3);
    }
    v16h a0 = load_a16(A, row0, k0, K);
    v16h a1 = load_a16(A, row0 + 16, k0, K);
#pragma unroll
    for (int t = 0; t < 4; ++t) {
      v16h b = load_b16(W, n0 + t * 16, k0, K);
      acc[0][t] = wmma_f16(a0, b, acc[0][t]);
      acc[1][t] = wmma_f16(a1, b, acc[1][t]);
    }
  }
  int l = laneid();
  int nn = l & 15;
#pragma unroll
  for (int r = 0; r < 2; ++r) {
    int mb = row0 + r * 16 + ((l & 16) ? 8 : 0);
#pragma unroll
    for (int t = 0; t < 4; ++t) {
      int col = n0 + t * 16 + nn;
      float bv = bias ? bias[col] : 0.0f;
      store_tile<MODE>(outp, acc[r][t], mb, col, N, MT, bv);
    }
  }
}

// ---------------------------------------------------------------------------
// Flash attention: per (window, head). Wave owns 16 q rows, streams 384 keys
// in tiles of 32. d = 32. grid.x = 36 * 4 * 12, block = 256 (8 waves).
// Q/K are f16 [token][128]; V is TRANSPOSED f16 [128][NKTOK] so the P@V
// B-fragment is two contiguous b128 loads per lane. Out f16 [token][128].
// ---------------------------------------------------------------------------
__global__ __launch_bounds__(256) void attn_k(const h16* __restrict__ Q,
                                              const h16* __restrict__ Kp,
                                              const h16* __restrict__ Vt,
                                              h16* __restrict__ Out) {
  __shared__ h16 lp[8][16][40];   // 40-half row stride -> 16B-aligned chunks
  int wave = threadIdx.x >> 5;
  int l = laneid();
  int bt = blockIdx.x % 12;
  int lh = blockIdx.x / 12;
  int h = lh & 3;
  int win = lh >> 2;
  int row0 = win * 1536 + (bt * 8 + wave) * 16;
  int kbase = win * 384;
  int co = h * DH;

  v16h qa = load_a16(Q, row0, co, CDIM);
  v8f o0 = {}, o1 = {};
  float m[8], ls[8];
#pragma unroll
  for (int v = 0; v < 8; ++v) { m[v] = -3.0e38f; ls[v] = 0.0f; }
  const float sc = 0.17677669529663687f; // 32^-0.5

  for (int kt = 0; kt < 384; kt += 32) {
    v16h kb0 = load_b16(Kp, kbase + kt, co, CDIM);       // S = Q @ K^T
    v16h kb1 = load_b16(Kp, kbase + kt + 16, co, CDIM);
    v8f z = {};
    v8f s0 = wmma_f16(qa, kb0, z);
    v8f s1 = wmma_f16(qa, kb1, z);
    int r8 = (l & 16) ? 8 : 0;
    int nn = l & 15;
#pragma unroll
    for (int v = 0; v < 8; ++v) {
      float x0 = s0[v] * sc, x1 = s1[v] * sc;
      float mx = fmaxf(x0, x1);
      mx = fmaxf(mx, __shfl_xor(mx, 1, 32));
      mx = fmaxf(mx, __shfl_xor(mx, 2, 32));
      mx = fmaxf(mx, __shfl_xor(mx, 4, 32));
      mx = fmaxf(mx, __shfl_xor(mx, 8, 32));
      float mn = fmaxf(m[v], mx);
      float alpha = __expf(m[v] - mn);
      float p0 = __expf(x0 - mn);
      float p1 = __expf(x1 - mn);
      float rs = p0 + p1;
      rs += __shfl_xor(rs, 1, 32);
      rs += __shfl_xor(rs, 2, 32);
      rs += __shfl_xor(rs, 4, 32);
      rs += __shfl_xor(rs, 8, 32);
      ls[v] = ls[v] * alpha + rs;
      m[v] = mn;
      o0[v] *= alpha;
      o1[v] *= alpha;
      lp[wave][v + r8][nn] = (h16)p0;        // C-layout -> LDS tile
      lp[wave][v + r8][nn + 16] = (h16)p1;
    }
    // LDS -> A-layout, two ds_load_b128 per lane
    {
      int hi = (l & 16) ? 8 : 0;
      const h16* base = &lp[wave][l & 15][hi];
      v16h pa = combine16(*(const v8h*)base, *(const v8h*)(base + 16));
      // V^T: B[k][n] = Vt[co+n][kbase+kt+k] -> contiguous along k
      v16h vb0 = load_b16(Vt, co, kbase + kt, NKTOK);     // O += P @ V
      v16h vb1 = load_b16(Vt, co + 16, kbase + kt, NKTOK);
      o0 = wmma_f16(pa, vb0, o0);
      o1 = wmma_f16(pa, vb1, o1);
    }
  }
  int nn = l & 15;
  int mb = row0 + ((l & 16) ? 8 : 0);
#pragma unroll
  for (int v = 0; v < 8; ++v) {
    float inv = 1.0f / ls[v];
    Out[(size_t)(mb + v) * CDIM + co + nn] = (h16)(o0[v] * inv);
    Out[(size_t)(mb + v) * CDIM + co + 16 + nn] = (h16)(o1[v] * inv);
  }
}

// ---------------------------------------------------------------------------
// Weight f32 -> f16 conversion (weights are tiny; one pass per matrix)
// ---------------------------------------------------------------------------
__global__ void cvt_k(const float* __restrict__ src, h16* __restrict__ dst, int n) {
  int t = blockIdx.x * blockDim.x + threadIdx.x;
  if (t < n) dst[t] = (h16)src[t];
}

// ---------------------------------------------------------------------------
// Geometry / elementwise kernels
// ---------------------------------------------------------------------------
__global__ void cam_embed_k(const float* __restrict__ I_inv,
                            const float* __restrict__ E_inv,
                            const float* __restrict__ W_img,
                            const float* __restrict__ W_cam,
                            float* __restrict__ img_tok) {
  int t = blockIdx.x * blockDim.x + threadIdx.x;
  if (t >= NCAM * HWF) return;
  int n = t / HWF, hw = t % HWF;
  int i = hw / FHW, j = hw % FHW;
  float px = j * (512.0f / 47.0f);
  float py = i * (512.0f / 47.0f);
  const float* Iv = I_inv + n * 9;
  float c0 = Iv[0] * px + Iv[1] * py + Iv[2];
  float c1 = Iv[3] * px + Iv[4] * py + Iv[5];
  float c2 = Iv[6] * px + Iv[7] * py + Iv[8];
  const float* Ev = E_inv + n * 16;
  float d4[4], e3[4];
#pragma unroll
  for (int r = 0; r < 4; ++r) {
    d4[r] = Ev[r * 4 + 0] * c0 + Ev[r * 4 + 1] * c1 + Ev[r * 4 + 2] * c2 + Ev[r * 4 + 3];
    e3[r] = Ev[r * 4 + 3];
  }
  float ss = 0.f;
  for (int c = 0; c < CDIM; ++c) {
    const float* wi = W_img + c * 4;
    const float* wc = W_cam + c * 4;
    float v = (wi[0] * d4[0] + wi[1] * d4[1] + wi[2] * d4[2] + wi[3] * d4[3]) -
              (wc[0] * e3[0] + wc[1] * e3[1] + wc[2] * e3[2] + wc[3] * e3[3]);
    ss += v * v;
  }
  float inv = 1.0f / (sqrtf(ss) + 1e-7f);
  float* dst = img_tok + (size_t)t * CDIM;
  for (int c = 0; c < CDIM; ++c) {
    const float* wi = W_img + c * 4;
    const float* wc = W_cam + c * 4;
    float v = (wi[0] * d4[0] + wi[1] * d4[1] + wi[2] * d4[2] + wi[3] * d4[3]) -
              (wc[0] * e3[0] + wc[1] * e3[1] + wc[2] * e3[2] + wc[3] * e3[3]);
    dst[c] = v * inv;
  }
}

__global__ void feat_relu_k(const float* __restrict__ feat,
                            const float* __restrict__ kg, const float* __restrict__ kb,
                            const float* __restrict__ vg, const float* __restrict__ vb,
                            h16* __restrict__ rk, h16* __restrict__ rv) {
  int t = blockIdx.x * blockDim.x + threadIdx.x;
  if (t >= NKTOK) return;
  int n = t / HWF, hw = t % HWF;
  const float bs = 0.99999500003749968f; // (1+1e-5)^-0.5
  for (int c = 0; c < CDIM; ++c) {
    float f = feat[((size_t)(n * CDIM + c)) * HWF + hw];
    float a = f * (kg[c] * bs) + kb[c];
    float b = f * (vg[c] * bs) + vb[c];
    rk[(size_t)t * CDIM + c] = (h16)fmaxf(a, 0.0f);
    rv[(size_t)t * CDIM + c] = (h16)fmaxf(b, 0.0f);
  }
}

// query = normalize(w_embed - c_embed) + x, then LN -> f16 tokens (attn1 q)
__global__ void build_q_k(const float* __restrict__ x, const float* __restrict__ E_inv,
                          const float* __restrict__ W_cam, const float* __restrict__ W_bev,
                          const float* __restrict__ b_bev, const float* __restrict__ g,
                          const float* __restrict__ b, h16* __restrict__ qout) {
  int t = blockIdx.x * blockDim.x + threadIdx.x;
  if (t >= NQTOK) return;
  int tt = t;
  int w2 = tt & 15; tt >>= 4;
  int w1 = tt & 15; tt >>= 4;
  int n = tt % 6; tt /= 6;
  int y = tt % 6; int xg = tt / 6;
  int hh = xg * 16 + w1, ww = y * 16 + w2;
  float g0 = (96.0f - hh * (192.0f / 95.0f)) * (1.0f / 1.92f);
  float g1 = (96.0f - ww * (192.0f / 95.0f)) * (1.0f / 1.92f);
  const float* Ev = E_inv + n * 16;
  float e3[4] = {Ev[3], Ev[7], Ev[11], Ev[15]};
  float ss = 0.f;
  for (int c = 0; c < CDIM; ++c) {
    const float* wc = W_cam + c * 4;
    float d = W_bev[c * 2] * g0 + W_bev[c * 2 + 1] * g1 + b_bev[c] -
              (wc[0] * e3[0] + wc[1] * e3[1] + wc[2] * e3[2] + wc[3] * e3[3]);
    ss += d * d;
  }
  float inv = 1.0f / (sqrtf(ss) + 1e-7f);
  int sp = hh * HB + ww;
  float sm = 0.f, sq = 0.f;
  for (int c = 0; c < CDIM; ++c) {
    const float* wc = W_cam + c * 4;
    float d = W_bev[c * 2] * g0 + W_bev[c * 2 + 1] * g1 + b_bev[c] -
              (wc[0] * e3[0] + wc[1] * e3[1] + wc[2] * e3[2] + wc[3] * e3[3]);
    float q = d * inv + x[(size_t)c * HWB + sp];
    sm += q; sq += q * q;
  }
  float mean = sm * (1.0f / CDIM);
  float var = sq * (1.0f / CDIM) - mean * mean;
  float rstd = rsqrtf(var + 1e-5f);
  h16* dst = qout + (size_t)t * CDIM;
  for (int c = 0; c < CDIM; ++c) {
    const float* wc = W_cam + c * 4;
    float d = W_bev[c * 2] * g0 + W_bev[c * 2 + 1] * g1 + b_bev[c] -
              (wc[0] * e3[0] + wc[1] * e3[1] + wc[2] * e3[2] + wc[3] * e3[3]);
    float q = d * inv + x[(size_t)c * HWB + sp];
    dst[c] = (h16)((q - mean) * rstd * g[c] + b[c]);
  }
}

// LN of key/val (spatial f32 [n*HWF][128]) into window(0)/grid(1) token order
__global__ void ln_kv_k(const float* __restrict__ src, const float* __restrict__ g,
                        const float* __restrict__ b, h16* __restrict__ dst, int gridmap) {
  int t = blockIdx.x * blockDim.x + threadIdx.x;
  if (t >= NKTOK) return;
  int tt = t;
  int w2 = tt & 7; tt >>= 3;
  int w1 = tt & 7; tt >>= 3;
  int n = tt % 6; tt /= 6;
  int y = tt % 6; int xg = tt / 6;
  int hh, ww;
  if (gridmap) { hh = w1 * 6 + xg; ww = w2 * 6 + y; }
  else         { hh = xg * 8 + w1; ww = y * 8 + w2; }
  const float* row = src + ((size_t)(n * HWF + hh * FHW + ww)) * CDIM;
  float sm = 0.f, sq = 0.f;
  for (int c = 0; c < CDIM; ++c) { float v = row[c]; sm += v; sq += v * v; }
  float mean = sm * (1.0f / CDIM);
  float rstd = rsqrtf(sq * (1.0f / CDIM) - mean * mean + 1e-5f);
  h16* d = dst + (size_t)t * CDIM;
  for (int c = 0; c < CDIM; ++c)
    d[c] = (h16)((row[c] - mean) * rstd * g[c] + b[c]);
}

// q2 tokens: broadcast x_skip[hw] across cameras, LN -> f16 (attn2 q)
__global__ void ln_q2_k(const float* __restrict__ src, const float* __restrict__ g,
                        const float* __restrict__ b, h16* __restrict__ dst) {
  int t = blockIdx.x * blockDim.x + threadIdx.x;
  if (t >= NQTOK) return;
  int tt = t;
  int w2 = tt & 15; tt >>= 4;
  int w1 = tt & 15; tt >>= 4;
  tt /= 6; // camera index (value identical across cams)
  int y = tt % 6; int xg = tt / 6;
  int sp = (xg * 16 + w1) * HB + (y * 16 + w2);
  const float* row = src + (size_t)sp * CDIM;
  float sm = 0.f, sq = 0.f;
  for (int c = 0; c < CDIM; ++c) { float v = row[c]; sm += v; sq += v * v; }
  float mean = sm * (1.0f / CDIM);
  float rstd = rsqrtf(sq * (1.0f / CDIM) - mean * mean + 1e-5f);
  h16* d = dst + (size_t)t * CDIM;
  for (int c = 0; c < CDIM; ++c)
    d[c] = (h16)((row[c] - mean) * rstd * g[c] + b[c]);
}

__global__ void ln_rows_k(const float* __restrict__ src, const float* __restrict__ g,
                          const float* __restrict__ b, h16* __restrict__ dst, int nrows) {
  int t = blockIdx.x * blockDim.x + threadIdx.x;
  if (t >= nrows) return;
  const float* row = src + (size_t)t * CDIM;
  float sm = 0.f, sq = 0.f;
  for (int c = 0; c < CDIM; ++c) { float v = row[c]; sm += v; sq += v * v; }
  float mean = sm * (1.0f / CDIM);
  float rstd = rsqrtf(sq * (1.0f / CDIM) - mean * mean + 1e-5f);
  h16* d = dst + (size_t)t * CDIM;
  for (int c = 0; c < CDIM; ++c)
    d[c] = (h16)((row[c] - mean) * rstd * g[c] + b[c]);
}

// mean over cameras of z tokens + skip; out [hw][c]
__global__ void reduce_skip_k(const float* __restrict__ z, const float* __restrict__ skip,
                              int skip_chw, float* __restrict__ out) {
  int t = blockIdx.x * blockDim.x + threadIdx.x;
  if (t >= HWB * CDIM) return;
  int c = t & 127;
  int hw = t >> 7;
  int hh = hw / HB, ww = hw % HB;
  int xg = hh >> 4, w1 = hh & 15, y = ww >> 4, w2 = ww & 15;
  size_t base = ((size_t)((xg * 6 + y) * 6)) * 256 + w1 * 16 + w2;
  float s = 0.f;
#pragma unroll
  for (int n = 0; n < NCAM; ++n) s += z[(base + (size_t)n * 256) * CDIM + c];
  float sk = skip_chw ? skip[(size_t)c * HWB + hw] : skip[(size_t)hw * CDIM + c];
  out[(size_t)hw * CDIM + c] = s * (1.0f / 6.0f) + sk;
}

__global__ void add_k(float* __restrict__ out, const float* __restrict__ a, int n) {
  int t = blockIdx.x * blockDim.x + threadIdx.x;
  if (t < n) out[t] += a[t];
}

__global__ void final_ln_k(const float* __restrict__ src, const float* __restrict__ g,
                           const float* __restrict__ b, float* __restrict__ out) {
  int t = blockIdx.x * blockDim.x + threadIdx.x;
  if (t >= HWB) return;
  const float* row = src + (size_t)t * CDIM;
  float sm = 0.f, sq = 0.f;
  for (int c = 0; c < CDIM; ++c) { float v = row[c]; sm += v; sq += v * v; }
  float mean = sm * (1.0f / CDIM);
  float rstd = rsqrtf(sq * (1.0f / CDIM) - mean * mean + 1e-5f);
  for (int c = 0; c < CDIM; ++c)
    out[(size_t)c * HWB + t] = (row[c] - mean) * rstd * g[c] + b[c];
}

// ---------------------------------------------------------------------------
// Launcher
// ---------------------------------------------------------------------------
extern "C" void kernel_launch(void* const* d_in, const int* in_sizes, int n_in,
                              void* d_out, int out_size, void* d_ws, size_t ws_size,
                              hipStream_t stream) {
  (void)in_sizes; (void)n_in; (void)out_size; (void)ws_size;
  const float* x       = (const float*)d_in[0];
  const float* feature = (const float*)d_in[1];
  const float* I_inv   = (const float*)d_in[2];
  const float* E_inv   = (const float*)d_in[3];
  int pi = 4;
  auto F = [&]() { return (const float*)d_in[pi++]; };
  const float* W_img = F();  const float* W_cam = F();
  const float* W_bev = F();  const float* b_bev = F();
  const float* W_kproj = F(); const float* W_vlin = F();
  const float* bnk_g = F(); const float* bnk_b = F();
  const float* bnv_g = F(); const float* bnv_b = F();
  struct AttnP { const float *lnq_g,*lnq_b,*lnk_g,*lnk_b,*lnv_g,*lnv_b,*Wq,*bq,*Wk,*bk,*Wv,*bv,*Wo,*bo; };
  auto loadAttn = [&](AttnP& a) {
    a.lnq_g=F(); a.lnq_b=F(); a.lnk_g=F(); a.lnk_b=F(); a.lnv_g=F(); a.lnv_b=F();
    a.Wq=F(); a.bq=F(); a.Wk=F(); a.bk=F(); a.Wv=F(); a.bv=F(); a.Wo=F(); a.bo=F();
  };
  AttnP A1, A2; loadAttn(A1); loadAttn(A2);
  struct MlpP { const float *W1,*b1,*W2,*b2; };
  MlpP M1, M2;
  M1.W1=F(); M1.b1=F(); M1.W2=F(); M1.b2=F();
  M2.W1=F(); M2.b1=F(); M2.W2=F(); M2.b2=F();
  const float* pre1_g=F(); const float* pre1_b=F();
  const float* pre2_g=F(); const float* pre2_b=F();
  const float* post_g=F(); const float* post_b=F();
  // d_in[pi] = index (unused; grid0 baked in)

  // workspace bump allocator (256-B aligned blocks)
  char* wp = (char*)d_ws;
  auto alloc = [&](size_t bytes) {
    void* r = (void*)wp;
    wp += (bytes + 255) & ~((size_t)255);
    return r;
  };
  float* img_tok = (float*)alloc((size_t)NKTOK * CDIM * 4);
  float* key_sp  = (float*)alloc((size_t)NKTOK * CDIM * 4);
  float* val_sp  = (float*)alloc((size_t)NKTOK * CDIM * 4);
  h16*   rk      = (h16*)  alloc((size_t)NKTOK * CDIM * 2);
  h16*   rv      = (h16*)  alloc((size_t)NKTOK * CDIM * 2);
  h16*   qln     = (h16*)  alloc((size_t)NQTOK * CDIM * 2);
  h16*   kln     = (h16*)  alloc((size_t)NKTOK * CDIM * 2);
  h16*   vln     = (h16*)  alloc((size_t)NKTOK * CDIM * 2);
  h16*   qp      = (h16*)  alloc((size_t)NQTOK * CDIM * 2);
  h16*   kp      = (h16*)  alloc((size_t)NKTOK * CDIM * 2);
  h16*   vpT     = (h16*)  alloc((size_t)CDIM * NKTOK * 2);  // V projected, transposed [d][token]
  h16*   aout    = (h16*)  alloc((size_t)NQTOK * CDIM * 2);
  float* zout    = (float*)alloc((size_t)NQTOK * CDIM * 4);
  float* qsp     = (float*)alloc((size_t)HWB * CDIM * 4);
  float* qsp2    = (float*)alloc((size_t)HWB * CDIM * 4);
  h16*   lnt     = (h16*)  alloc((size_t)HWB * CDIM * 2);
  h16*   hid     = (h16*)  alloc((size_t)HWB * 2 * CDIM * 2);
  float* mlpo    = (float*)alloc((size_t)HWB * CDIM * 4);
  // f16 weight arena
  const int WSM = CDIM * CDIM;       // 16384
  const int WLG = 2 * CDIM * CDIM;   // 32768
  h16* kprj_h = (h16*)alloc(WSM * 2);
  h16* vlin_h = (h16*)alloc(WSM * 2);
  h16* wq1_h  = (h16*)alloc(WSM * 2);
  h16* wk1_h  = (h16*)alloc(WSM * 2);
  h16* wv1_h  = (h16*)alloc(WSM * 2);
  h16* wo1_h  = (h16*)alloc(WSM * 2);
  h16* wq2_h  = (h16*)alloc(WSM * 2);
  h16* wk2_h  = (h16*)alloc(WSM * 2);
  h16* wv2_h  = (h16*)alloc(WSM * 2);
  h16* wo2_h  = (h16*)alloc(WSM * 2);
  h16* w11_h  = (h16*)alloc(WLG * 2);
  h16* w12_h  = (h16*)alloc(WLG * 2);
  h16* w21_h  = (h16*)alloc(WLG * 2);
  h16* w22_h  = (h16*)alloc(WLG * 2);

  dim3 B(256);
  // --- weight conversions (tiny) ---
  cvt_k<<<WSM / 256, B, 0, stream>>>(W_kproj, kprj_h, WSM);
  cvt_k<<<WSM / 256, B, 0, stream>>>(W_vlin, vlin_h, WSM);
  cvt_k<<<WSM / 256, B, 0, stream>>>(A1.Wq, wq1_h, WSM);
  cvt_k<<<WSM / 256, B, 0, stream>>>(A1.Wk, wk1_h, WSM);
  cvt_k<<<WSM / 256, B, 0, stream>>>(A1.Wv, wv1_h, WSM);
  cvt_k<<<WSM / 256, B, 0, stream>>>(A1.Wo, wo1_h, WSM);
  cvt_k<<<WSM / 256, B, 0, stream>>>(A2.Wq, wq2_h, WSM);
  cvt_k<<<WSM / 256, B, 0, stream>>>(A2.Wk, wk2_h, WSM);
  cvt_k<<<WSM / 256, B, 0, stream>>>(A2.Wv, wv2_h, WSM);
  cvt_k<<<WSM / 256, B, 0, stream>>>(A2.Wo, wo2_h, WSM);
  cvt_k<<<WLG / 256, B, 0, stream>>>(M1.W1, w11_h, WLG);
  cvt_k<<<WLG / 256, B, 0, stream>>>(M1.W2, w12_h, WLG);
  cvt_k<<<WLG / 256, B, 0, stream>>>(M2.W1, w21_h, WLG);
  cvt_k<<<WLG / 256, B, 0, stream>>>(M2.W2, w22_h, WLG);

  // --- shared geometry + key/value features ---
  cam_embed_k<<<NKTOK / 256, B, 0, stream>>>(I_inv, E_inv, W_img, W_cam, img_tok);
  feat_relu_k<<<NKTOK / 256, B, 0, stream>>>(feature, bnk_g, bnk_b, bnv_g, bnv_b, rk, rv);
  gemm_k<0><<<dim3(NKTOK / 256, 2), B, 0, stream>>>(rk, kprj_h, nullptr, key_sp, CDIM, CDIM, 0);
  gemm_k<0><<<dim3(NKTOK / 256, 2), B, 0, stream>>>(rv, vlin_h, nullptr, val_sp, CDIM, CDIM, 0);
  add_k<<<(NKTOK * CDIM) / 256, B, 0, stream>>>(key_sp, img_tok, NKTOK * CDIM);

  // --- attention 1 (local-to-local, window partition) ---
  build_q_k<<<NQTOK / 256, B, 0, stream>>>(x, E_inv, W_cam, W_bev, b_bev, A1.lnq_g, A1.lnq_b, qln);
  ln_kv_k<<<NKTOK / 256, B, 0, stream>>>(key_sp, A1.lnk_g, A1.lnk_b, kln, 0);
  ln_kv_k<<<NKTOK / 256, B, 0, stream>>>(val_sp, A1.lnv_g, A1.lnv_b, vln, 0);
  gemm_k<1><<<dim3(NQTOK / 256, 2), B, 0, stream>>>(qln, wq1_h, A1.bq, qp, CDIM, CDIM, 0);
  gemm_k<1><<<dim3(NKTOK / 256, 2), B, 0, stream>>>(kln, wk1_h, A1.bk, kp, CDIM, CDIM, 0);
  gemm_k<3><<<dim3(NKTOK / 256, 2), B, 0, stream>>>(vln, wv1_h, A1.bv, vpT, CDIM, CDIM, NKTOK);
  attn_k<<<36 * HEADS * 12, B, 0, stream>>>(qp, kp, vpT, aout);
  gemm_k<0><<<dim3(NQTOK / 256, 2), B, 0, stream>>>(aout, wo1_h, A1.bo, zout, CDIM, CDIM, 0);
  reduce_skip_k<<<(HWB * CDIM) / 256, B, 0, stream>>>(zout, x, 1, qsp);
  // MLP 1
  ln_rows_k<<<HWB / 256, B, 0, stream>>>(qsp, pre1_g, pre1_b, lnt, HWB);
  gemm_k<2><<<dim3(HWB / 256, 4), B, 0, stream>>>(lnt, w11_h, M1.b1, hid, 2 * CDIM, CDIM, 0);
  gemm_k<0><<<dim3(HWB / 256, 2), B, 0, stream>>>(hid, w12_h, M1.b2, mlpo, CDIM, 2 * CDIM, 0);
  add_k<<<(HWB * CDIM) / 256, B, 0, stream>>>(qsp, mlpo, HWB * CDIM);

  // --- attention 2 (local-to-global, grid partition on key/value) ---
  ln_q2_k<<<NQTOK / 256, B, 0, stream>>>(qsp, A2.lnq_g, A2.lnq_b, qln);
  ln_kv_k<<<NKTOK / 256, B, 0, stream>>>(key_sp, A2.lnk_g, A2.lnk_b, kln, 1);
  ln_kv_k<<<NKTOK / 256, B, 0, stream>>>(val_sp, A2.lnv_g, A2.lnv_b, vln, 1);
  gemm_k<1><<<dim3(NQTOK / 256, 2), B, 0, stream>>>(qln, wq2_h, A2.bq, qp, CDIM, CDIM, 0);
  gemm_k<1><<<dim3(NKTOK / 256, 2), B, 0, stream>>>(kln, wk2_h, A2.bk, kp, CDIM, CDIM, 0);
  gemm_k<3><<<dim3(NKTOK / 256, 2), B, 0, stream>>>(vln, wv2_h, A2.bv, vpT, CDIM, CDIM, NKTOK);
  attn_k<<<36 * HEADS * 12, B, 0, stream>>>(qp, kp, vpT, aout);
  gemm_k<0><<<dim3(NQTOK / 256, 2), B, 0, stream>>>(aout, wo2_h, A2.bo, zout, CDIM, CDIM, 0);
  reduce_skip_k<<<(HWB * CDIM) / 256, B, 0, stream>>>(zout, qsp, 0, qsp2);
  // MLP 2
  ln_rows_k<<<HWB / 256, B, 0, stream>>>(qsp2, pre2_g, pre2_b, lnt, HWB);
  gemm_k<2><<<dim3(HWB / 256, 4), B, 0, stream>>>(lnt, w21_h, M2.b1, hid, 2 * CDIM, CDIM, 0);
  gemm_k<0><<<dim3(HWB / 256, 2), B, 0, stream>>>(hid, w22_h, M2.b2, mlpo, CDIM, 2 * CDIM, 0);
  add_k<<<(HWB * CDIM) / 256, B, 0, stream>>>(qsp2, mlpo, HWB * CDIM);

  // --- final LN + transpose to (b, dim, H, W) ---
  final_ln_k<<<HWB / 256, B, 0, stream>>>(qsp2, post_g, post_b, (float*)d_out);
}
